// StandardTransformerLayer_11888469475632
// MI455X (gfx1250) — matrix operand order repaired and checked by reading, not compiled, not run on hardware
//
#include <hip/hip_runtime.h>
#include <hip/hip_bf16.h>
#include <math.h>

// MI455X / gfx1250, wave32. GEMMs + attention on v_wmma_f32_16x16x32_f16,
// A-operand staging via ASYNCcnt-tracked global_load_async_to_lds_b128.

typedef __attribute__((ext_vector_type(16))) _Float16 v16h;
typedef __attribute__((ext_vector_type(8)))  _Float16 v8h;
typedef __attribute__((ext_vector_type(8)))  float    v8f;

#define DEVINL __device__ __forceinline__

// A fragment (16x32 f16, M = lane&15): lane holds row M,
// elements e -> K = 16*half + e. Two 16B loads.
// Works for global or LDS source (compiler picks vmem/ds by address space).
DEVINL v16h load_a_frag(const _Float16* __restrict__ p, int lda) {
  const int lane = threadIdx.x & 31;
  const int r = lane & 15, hf = lane >> 4;
  const _Float16* rp = p + (size_t)r * lda + 16 * hf;
  v8h lo = *(const v8h*)rp;
  v8h hi = *(const v8h*)(rp + 8);
  v16h out;
#pragma unroll
  for (int i = 0; i < 8; ++i) { out[i] = lo[i]; out[i + 8] = hi[i]; }
  return out;
}

// B fragment (32x16 f16): lane holds output column N = lane&15 (== weight row),
// elements e -> K = 16*half + e. One contiguous 32B load from row-major [N,K].
DEVINL v16h load_b_frag(const _Float16* __restrict__ p, int ldb) {
  const int lane = threadIdx.x & 31;
  const int r = lane & 15, hf = lane >> 4;
  return *(const v16h*)(p + (size_t)r * ldb + 16 * hf);
}

DEVINL v8f wmma_f16(v16h a, v16h b, v8f c) {
  return __builtin_amdgcn_wmma_f32_16x16x32_f16(false, a, false, b, (short)0, c,
                                                false, false);
}

// CDNA5 async global->LDS copy (ASYNCcnt). VDST = LDS byte address, GV mode.
DEVINL void async_ld_b128(unsigned lds_off, const _Float16* g) {
  unsigned long long ga = (unsigned long long)(uintptr_t)g;
  asm volatile("global_load_async_to_lds_b128 %0, %1, off"
               : : "v"(lds_off), "v"(ga) : "memory");
}
DEVINL void wait_async0() {
  asm volatile("s_wait_asynccnt 0" : : : "memory");
}

// ---------------------------------------------------------------- utilities
__global__ __launch_bounds__(256) void k_f32_to_f16(
    const float* __restrict__ in, _Float16* __restrict__ out, int n) {
  int i = blockIdx.x * 256 + threadIdx.x;
  const int stride = gridDim.x * 256;
  for (; i < n; i += stride) out[i] = (_Float16)in[i];
}

__global__ __launch_bounds__(256) void k_layernorm_f16(
    const float* __restrict__ x, const float* __restrict__ g,
    const float* __restrict__ b, _Float16* __restrict__ out, int C) {
  const int row = blockIdx.x;
  const float* xr = x + (size_t)row * C;
  float s = 0.f, ss = 0.f;
  for (int i = threadIdx.x; i < C; i += 256) { float v = xr[i]; s += v; ss += v * v; }
#pragma unroll
  for (int m = 16; m >= 1; m >>= 1) { s += __shfl_xor(s, m, 32); ss += __shfl_xor(ss, m, 32); }
  __shared__ float red[16];
  const int w = threadIdx.x >> 5;
  if ((threadIdx.x & 31) == 0) { red[w] = s; red[8 + w] = ss; }
  __syncthreads();
  float S = 0.f, SS = 0.f;
#pragma unroll
  for (int i = 0; i < 8; ++i) { S += red[i]; SS += red[8 + i]; }
  const float mu = S / C;
  const float rstd = rsqrtf(SS / C - mu * mu + 1e-5f);
  _Float16* orow = out + (size_t)row * C;
  for (int i = threadIdx.x; i < C; i += 256)
    orow[i] = (_Float16)((xr[i] - mu) * rstd * g[i] + b[i]);
}

// ------------------------------------------------------------- WMMA GEMM
// out[M,N] = A[M,K] (f16) * Bw[N,K]^T (f16), f32 accumulate.
// Block: 256 thr = 8 waves as 2(M) x 4(N); wave tile 32x128; block tile 64x512.
// A chunk (64x32) double-buffered in LDS via async DMA, issued 1 iter ahead.
template <int RESID, int GELU, int OUTF32, int OUTF16>
__global__ __launch_bounds__(256) void k_gemm_wmma(
    const _Float16* __restrict__ A, const _Float16* __restrict__ Bw,
    const float* __restrict__ resid, float* __restrict__ Cf,
    _Float16* __restrict__ Ch, int M, int N, int K) {
  const int wave = threadIdx.x >> 5;
  const int wm = wave & 1, wn = wave >> 1;
  const int mblk = blockIdx.x * 64;
  const int m0 = mblk + wm * 32;
  const int n0 = blockIdx.y * 512 + wn * 128;
  const int lane = threadIdx.x & 31;
  const int r = lane & 15, hf = lane >> 4;

  __shared__ _Float16 As[2][64][32];  // 8KB double buffer

  v8f zero = {};
  v8f acc[2][8];
#pragma unroll
  for (int i = 0; i < 2; ++i)
#pragma unroll
    for (int j = 0; j < 8; ++j) acc[i][j] = zero;

  // one b128 async load per thread per chunk: 64 rows x 64B = 4KB
  const int srow = threadIdx.x >> 2, sseg = threadIdx.x & 3;
  const _Float16* sgp = A + (size_t)(mblk + srow) * K + sseg * 8;

  const int nk = K >> 5;
  async_ld_b128((unsigned)(uintptr_t)&As[0][srow][sseg * 8], sgp);  // k = 0

  for (int kc = 0; kc < nk; ++kc) {
    const int cur = kc & 1;
    const int k = kc << 5;
    wait_async0();     // our buf[cur] copy has landed
    __syncthreads();   // everyone's copies landed; prior readers done
    if (kc + 1 < nk)   // prefetch next chunk into the other buffer
      async_ld_b128((unsigned)(uintptr_t)&As[cur ^ 1][srow][sseg * 8],
                    sgp + (k + 32));

    v16h a0 = load_a_frag(&As[cur][wm * 32 + 0][0], 32);
    v16h a1 = load_a_frag(&As[cur][wm * 32 + 16][0], 32);
#pragma unroll
    for (int j = 0; j < 8; ++j) {
      const _Float16* bp = Bw + (size_t)(n0 + 16 * j) * K + k;
      v16h b = load_b_frag(bp, K);
      if (k + 256 < K)  // gfx1250 global_prefetch_b8 hint
        __builtin_prefetch(bp + (size_t)r * K + 256, 0, 1);
      acc[0][j] = wmma_f16(a0, b, acc[0][j]);
      acc[1][j] = wmma_f16(a1, b, acc[1][j]);
    }
  }

#pragma unroll
  for (int i = 0; i < 2; ++i) {
#pragma unroll
    for (int j = 0; j < 8; ++j) {
#pragma unroll
      for (int e = 0; e < 8; ++e) {
        const int row = m0 + 16 * i + e + 8 * hf;  // C-layout: M = e + 8*half
        const int col = n0 + 16 * j + r;           // N = lane&15
        float v = acc[i][j][e];
        if constexpr (GELU) v = 0.5f * v * (1.0f + erff(v * 0.70710678118654752f));
        const size_t idx = (size_t)row * N + col;
        if constexpr (RESID) v += resid[idx];
        if constexpr (OUTF32) Cf[idx] = v;
        if constexpr (OUTF16) Ch[idx] = (_Float16)v;
      }
    }
  }
}

// -------------------------------------------------------- Flash attention
// qkv16: [B,T, 3*H*hd] f16 row-major (GEMM output layout). Per block:
// one (b, h, 64-query) tile, 4 waves x 16 rows. Online softmax, f32 O acc.
__global__ __launch_bounds__(128) void k_attn_flash(
    const _Float16* __restrict__ qkv, _Float16* __restrict__ o16, int T) {
  constexpr int HD = 128, NH = 16, D3 = 3 * NH * HD;  // 6144
  constexpr int CDIM = NH * HD;                       // 2048
  const int b = blockIdx.z, h = blockIdx.y;
  const int q0 = blockIdx.x * 64;
  const int wave = threadIdx.x >> 5;
  const int lane = threadIdx.x & 31;
  const int r = lane & 15, hf = lane >> 4;
  const _Float16* base = qkv + (size_t)b * T * D3;
  const int qoff = h * HD, koff = (NH + h) * HD, voff = (2 * NH + h) * HD;
  const int qrow = q0 + wave * 16;
  const float scale = 0.088388347648318447f;  // 1/sqrt(128)

  v16h qa[4];  // Q tile 16x128 resident in registers
#pragma unroll
  for (int d = 0; d < 4; ++d)
    qa[d] = load_a_frag(base + (size_t)qrow * D3 + qoff + 32 * d, D3);

  v8f zero = {};
  v8f oacc[8];  // 16x128 f32
#pragma unroll
  for (int n = 0; n < 8; ++n) oacc[n] = zero;
  float mrow[8], lrow[8];
#pragma unroll
  for (int e = 0; e < 8; ++e) { mrow[e] = -1e30f; lrow[e] = 0.f; }

  __shared__ _Float16 Vs[128][32];      // V chunk transposed: [hd][key], 8KB
  __shared__ _Float16 Ps[4][16][32];    // per-wave P relayout C->A, 4KB

  for (int kv = 0; kv < T; kv += 32) {
    __syncthreads();
    {  // stage V chunk (32 keys x 128 hd) transposed into LDS
      const int key = threadIdx.x >> 2, seg = threadIdx.x & 3;
      const _Float16* vp = base + (size_t)(kv + key) * D3 + voff + seg * 32;
#pragma unroll
      for (int u = 0; u < 32; ++u) Vs[seg * 32 + u][key] = vp[u];
    }
    __syncthreads();

    // S(16x32) = Q(16x128) * K(32x128)^T, two 16-key column tiles
    v8f s0a = zero, s1a = zero;
#pragma unroll
    for (int d = 0; d < 4; ++d) {
      v16h kb0 = load_b_frag(base + (size_t)kv * D3 + koff + 32 * d, D3);
      v16h kb1 = load_b_frag(base + (size_t)(kv + 16) * D3 + koff + 32 * d, D3);
      s0a = wmma_f16(qa[d], kb0, s0a);
      s1a = wmma_f16(qa[d], kb1, s1a);
    }

    // online softmax over rows; row M = e + 8*half lives in 16-lane halves
    float corr[8];
#pragma unroll
    for (int e = 0; e < 8; ++e) {
      float s0 = s0a[e] * scale, s1 = s1a[e] * scale;
      float mx = fmaxf(s0, s1);
#pragma unroll
      for (int m = 1; m < 16; m <<= 1) mx = fmaxf(mx, __shfl_xor(mx, m, 32));
      const float mn = fmaxf(mrow[e], mx);
      const float p0 = __expf(s0 - mn), p1 = __expf(s1 - mn);
      float ps = p0 + p1;
#pragma unroll
      for (int m = 1; m < 16; m <<= 1) ps += __shfl_xor(ps, m, 32);
      const float c = __expf(mrow[e] - mn);
      lrow[e] = lrow[e] * c + ps;
      mrow[e] = mn;
      corr[e] = c;
      Ps[wave][e + 8 * hf][r] = (_Float16)p0;       // C-layout -> LDS
      Ps[wave][e + 8 * hf][16 + r] = (_Float16)p1;
    }
#pragma unroll
    for (int n = 0; n < 8; ++n)
#pragma unroll
      for (int e = 0; e < 8; ++e) oacc[n][e] *= corr[e];

    // reload P as an A fragment (same-wave LDS ops are in-order)
    v16h pa;
    {
      const _Float16* rp = &Ps[wave][r][16 * hf];
      v8h lo = *(const v8h*)rp;
      v8h hi = *(const v8h*)(rp + 8);
#pragma unroll
      for (int i = 0; i < 8; ++i) { pa[i] = lo[i]; pa[i + 8] = hi[i]; }
    }
    // O += P(16x32) * V(32x128): B operand from transposed LDS V
#pragma unroll
    for (int n = 0; n < 8; ++n) {
      v16h vb = *(const v16h*)(&Vs[n * 16 + r][16 * hf]);
      oacc[n] = wmma_f16(pa, vb, oacc[n]);
    }
  }

  float inv[8];
#pragma unroll
  for (int e = 0; e < 8; ++e) inv[e] = 1.0f / lrow[e];
  // write O directly in (B, T, H*hd) layout for the projection GEMM
  _Float16* orow = o16 + ((size_t)b * T + qrow) * CDIM + h * HD;
#pragma unroll
  for (int n = 0; n < 8; ++n)
#pragma unroll
    for (int e = 0; e < 8; ++e)
      orow[(size_t)(e + 8 * hf) * CDIM + n * 16 + r] =
          (_Float16)(oacc[n][e] * inv[e]);
}

// ------------------------------------------------------------------ driver
extern "C" void kernel_launch(void* const* d_in, const int* in_sizes, int n_in,
                              void* d_out, int out_size, void* d_ws, size_t ws_size,
                              hipStream_t stream) {
  (void)in_sizes; (void)n_in; (void)out_size; (void)ws_size;
  const int B = 2, T = 2048, C = 2048;
  const int M = B * T;      // 4096
  const int C3 = 3 * C;     // 6144
  const int F = 4 * C;      // 8192

  const float* x      = (const float*)d_in[0];
  const float* g1     = (const float*)d_in[1];
  const float* b1     = (const float*)d_in[2];
  const float* g2     = (const float*)d_in[3];
  const float* b2     = (const float*)d_in[4];
  const float* w_qkv  = (const float*)d_in[5];
  const float* w_proj = (const float*)d_in[6];
  const float* w_fc1  = (const float*)d_in[7];
  const float* w_fc2  = (const float*)d_in[8];
  float* out = (float*)d_out;   // also serves as the fp32 residual stream

  _Float16* p = (_Float16*)d_ws;
  _Float16* wq16  = p; p += (size_t)C3 * C;
  _Float16* wp16  = p; p += (size_t)C * C;
  _Float16* wf1   = p; p += (size_t)F * C;
  _Float16* wf2   = p; p += (size_t)C * F;
  _Float16* h16   = p; p += (size_t)M * C;
  _Float16* qkv16 = p; p += (size_t)M * C3;
  _Float16* o16   = p; p += (size_t)M * C;
  _Float16* m16   = p; p += (size_t)M * F;

  // weights -> f16 (recomputed each launch; no cross-call state)
  k_f32_to_f16<<<4096, 256, 0, stream>>>(w_qkv,  wq16, C3 * C);
  k_f32_to_f16<<<2048, 256, 0, stream>>>(w_proj, wp16, C * C);
  k_f32_to_f16<<<4096, 256, 0, stream>>>(w_fc1,  wf1,  F * C);
  k_f32_to_f16<<<4096, 256, 0, stream>>>(w_fc2,  wf2,  C * F);

  // LN1 -> h16
  k_layernorm_f16<<<M, 256, 0, stream>>>(x, g1, b1, h16, C);

  // qkv = h @ w_qkv^T  (f16 out for attention)
  dim3 gqkv(M / 64, C3 / 512);
  k_gemm_wmma<0,0,0,1><<<gqkv, 256, 0, stream>>>(h16, wq16, nullptr, nullptr,
                                                 qkv16, M, C3, C);
  // attention -> o16 in (B,T,H*hd)
  dim3 gattn(T / 64, 16, B);
  k_attn_flash<<<gattn, 128, 0, stream>>>(qkv16, o16, T);

  // out = x + o @ w_proj^T   (f32 residual stream)
  dim3 gproj(M / 64, C / 512);
  k_gemm_wmma<1,0,1,0><<<gproj, 256, 0, stream>>>(o16, wp16, x, out, nullptr,
                                                  M, C, C);
  // LN2 -> h16 (reuse)
  k_layernorm_f16<<<M, 256, 0, stream>>>(out, g2, b2, h16, C);

  // m = gelu(h2 @ w_fc1^T)  (f16 out)
  dim3 gfc1(M / 64, F / 512);
  k_gemm_wmma<0,1,0,1><<<gfc1, 256, 0, stream>>>(h16, wf1, nullptr, nullptr,
                                                 m16, M, F, C);
  // out += m @ w_fc2^T
  dim3 gfc2(M / 64, C / 512);
  k_gemm_wmma<1,0,1,0><<<gfc2, 256, 0, stream>>>(m16, wf2, out, out, nullptr,
                                                 M, C, F);
}
